// Pointnet2Backbone_33870112096421
// MI455X (gfx1250) — matrix operand order repaired and checked by reading, not compiled, not run on hardware
//
#include <hip/hip_runtime.h>
#include <hip/hip_bf16.h>

typedef __attribute__((ext_vector_type(16))) _Float16 v16h;
typedef __attribute__((ext_vector_type(8)))  float    v8f;

#define BATCH 4
#define FPS_THREADS 256
#define GEMM_MB 4          // 16-row tiles per wave (64 rows/wave, 256 rows/block)

// ------------------------------------------------------------------
// Farthest point sampling: one block per batch. Sequential argmax over
// the running min-distance array, tie-break = lowest index (jnp.argmax).
// ------------------------------------------------------------------
__global__ void fps_kernel(const float* __restrict__ xyz, int N, int npoint,
                           int* __restrict__ inds, float* __restrict__ mind_ws) {
  const int b = blockIdx.x;
  const int t = threadIdx.x;
  const float* p = xyz + (size_t)b * N * 3;
  float* mind = mind_ws + (size_t)b * N;
  __shared__ float svals[FPS_THREADS];
  __shared__ int   sidx[FPS_THREADS];
  __shared__ float cur[3];
  if (t == 0) {
    inds[b * npoint] = 0;
    cur[0] = p[0]; cur[1] = p[1]; cur[2] = p[2];
  }
  __syncthreads();
  for (int i = t; i < N; i += FPS_THREADS) {
    float dx = p[i*3+0]-cur[0], dy = p[i*3+1]-cur[1], dz = p[i*3+2]-cur[2];
    mind[i] = dx*dx + dy*dy + dz*dz;
  }
  __syncthreads();
  for (int k = 1; k < npoint; ++k) {
    float bv = -1.0f; int bi = 0;
    for (int i = t; i < N; i += FPS_THREADS) {
      float v = mind[i];
      if (v > bv) { bv = v; bi = i; }
    }
    svals[t] = bv; sidx[t] = bi;
    __syncthreads();
    for (int s = FPS_THREADS / 2; s > 0; s >>= 1) {
      if (t < s) {
        float ov = svals[t + s]; int oi = sidx[t + s];
        if (ov > svals[t] || (ov == svals[t] && oi < sidx[t])) {
          svals[t] = ov; sidx[t] = oi;
        }
      }
      __syncthreads();
    }
    int sel = sidx[0];
    if (t == 0) {
      inds[b * npoint + k] = sel;
      cur[0] = p[sel*3+0]; cur[1] = p[sel*3+1]; cur[2] = p[sel*3+2];
    }
    __syncthreads();
    for (int i = t; i < N; i += FPS_THREADS) {
      float dx = p[i*3+0]-cur[0], dy = p[i*3+1]-cur[1], dz = p[i*3+2]-cur[2];
      float d = dx*dx + dy*dy + dz*dz;
      if (d < mind[i]) mind[i] = d;
    }
    __syncthreads();
  }
}

// ------------------------------------------------------------------
__global__ void gather_xyz_kernel(const float* __restrict__ xyz, const int* __restrict__ inds,
                                  float* __restrict__ out, int S, int N) {
  int tid = blockIdx.x * blockDim.x + threadIdx.x;
  if (tid >= BATCH * S) return;
  int b = tid / S;
  int j = inds[tid];
  const float* src = xyz + ((size_t)b * N + j) * 3;
  out[tid*3+0] = src[0]; out[tid*3+1] = src[1]; out[tid*3+2] = src[2];
}

// ------------------------------------------------------------------
// Ball query: first nsample points (ascending index) with d2 < r2;
// pad with first hit. One thread per query center.
// ------------------------------------------------------------------
__global__ void ball_query_kernel(const float* __restrict__ xyz, const float* __restrict__ new_xyz,
                                  int N, int S, float r2, int nsample, int* __restrict__ idx) {
  int tid = blockIdx.x * blockDim.x + threadIdx.x;
  if (tid >= BATCH * S) return;
  int b = tid / S;
  const float* p = xyz + (size_t)b * N * 3;
  float cx = new_xyz[tid*3+0], cy = new_xyz[tid*3+1], cz = new_xyz[tid*3+2];
  int* out = idx + (size_t)tid * nsample;
  int cnt = 0;
  for (int j = 0; j < N && cnt < nsample; ++j) {
    float dx = p[j*3+0]-cx, dy = p[j*3+1]-cy, dz = p[j*3+2]-cz;
    if (dx*dx + dy*dy + dz*dz < r2) out[cnt++] = j;
  }
  int first = (cnt > 0) ? out[0] : 0;
  for (; cnt < nsample; ++cnt) out[cnt] = first;
}

// ------------------------------------------------------------------
// Build grouped feature matrix X (M=B*S*nsample rows, Kpad cols, f16):
// cols 0..2 = (xyz[j]-center)/radius, then C feature cols, zero pad.
// ------------------------------------------------------------------
__global__ void group_kernel(const float* __restrict__ xyz, const float* __restrict__ new_xyz,
                             const _Float16* __restrict__ feats, int C, int N,
                             const int* __restrict__ idx, int S, int nsample,
                             float inv_r, int Kpad, _Float16* __restrict__ X) {
  int row = blockIdx.x * blockDim.x + threadIdx.x;
  int total = BATCH * S * nsample;
  if (row >= total) return;
  int k  = row % nsample;
  int bs = row / nsample;
  int b  = bs / S;
  int j  = idx[(size_t)bs * nsample + k];
  const float* pj = xyz + ((size_t)b * N + j) * 3;
  const float* c  = new_xyz + (size_t)bs * 3;
  _Float16* xr = X + (size_t)row * Kpad;
  xr[0] = (_Float16)((pj[0]-c[0]) * inv_r);
  xr[1] = (_Float16)((pj[1]-c[1]) * inv_r);
  xr[2] = (_Float16)((pj[2]-c[2]) * inv_r);
  int col = 3;
  if (feats) {
    const _Float16* f = feats + ((size_t)b * N + j) * C;
    for (int ci = 0; ci < C; ++ci) xr[col++] = f[ci];
  }
  for (; col < Kpad; ++col) xr[col] = (_Float16)0.0f;
}

// ------------------------------------------------------------------
// Pack weights: W (Cin x Cout, f32, row-major) -> Wt (Cout x Kpad, f16,
// transposed, zero-padded K). Makes the WMMA B-fragment load fully
// contiguous per lane (2x global_load_b128), no guards in the hot loop.
// ------------------------------------------------------------------
__global__ void pack_weights_kernel(const float* __restrict__ W, int Cin, int Cout, int Kpad,
                                    _Float16* __restrict__ Wt) {
  int tid = blockIdx.x * blockDim.x + threadIdx.x;
  if (tid >= Kpad * Cout) return;
  int k  = tid % Kpad;
  int nn = tid / Kpad;
  Wt[(size_t)nn * Kpad + k] = (k < Cin) ? (_Float16)W[(size_t)k * Cout + nn]
                                        : (_Float16)0.0f;
}

// ------------------------------------------------------------------
// Y = relu(X @ W + bias). X: (M x Kpad) f16 row-major (zero-padded K),
// Wt: (Cout x Kpad) f16 packed. One wave computes GEMM_MB stacked 16x16
// tiles (B fragment reused across 4 WMMAs per K step).
// Fragment packing per CDNA5 ISA 16-bit A (16x32) / B (32x16) layouts.
// ------------------------------------------------------------------
__global__ void gemm_bias_relu_wmma(const _Float16* __restrict__ X, int M, int Kpad,
                                    const _Float16* __restrict__ Wt,
                                    const float* __restrict__ bias, int Cout,
                                    _Float16* __restrict__ Yh, float* __restrict__ Yf) {
  const int wave  = threadIdx.x >> 5;
  const int lane  = threadIdx.x & 31;
  const int mBase = (blockIdx.x * (blockDim.x >> 5) + wave) * (16 * GEMM_MB);
  const int nTile = blockIdx.y;
  if (mBase >= M) return;                 // wave-uniform: EXEC stays all-ones
  const int hi   = lane >> 4;             // half-wave select
  const int mloc = lane & 15;
  const int n    = nTile * 16 + (lane & 15);
  const size_t rowStride = (size_t)16 * Kpad;

  v8f acc0 = {}, acc1 = {}, acc2 = {}, acc3 = {};
  const _Float16* bp = Wt + (size_t)n * Kpad + hi * 8;           // two runs of 8 halves (k=8h.., 16+8h..)
  const _Float16* ap = X + (size_t)(mBase + mloc) * Kpad + hi * 8;

#pragma unroll 1
  for (int k0 = 0; k0 < Kpad; k0 += 32) {
    v16h b;
#pragma unroll
    for (int e = 0; e < 8; ++e) { b[e] = bp[e]; b[8 + e] = bp[16 + e]; }
    __builtin_prefetch((const void*)(bp + 32), 0, 1);            // global_prefetch_b8
    bp += 32;

    v16h a0, a1, a2, a3;
    const _Float16* ar = ap;
#pragma unroll
    for (int e = 0; e < 8; ++e) { a0[e] = ar[e]; a0[8 + e] = ar[16 + e]; }
    ar += rowStride;
#pragma unroll
    for (int e = 0; e < 8; ++e) { a1[e] = ar[e]; a1[8 + e] = ar[16 + e]; }
    ar += rowStride;
#pragma unroll
    for (int e = 0; e < 8; ++e) { a2[e] = ar[e]; a2[8 + e] = ar[16 + e]; }
    ar += rowStride;
#pragma unroll
    for (int e = 0; e < 8; ++e) { a3[e] = ar[e]; a3[8 + e] = ar[16 + e]; }
    ap += 32;

    acc0 = __builtin_amdgcn_wmma_f32_16x16x32_f16(false, a0, false, b, (short)0, acc0, false, false);
    acc1 = __builtin_amdgcn_wmma_f32_16x16x32_f16(false, a1, false, b, (short)0, acc1, false, false);
    acc2 = __builtin_amdgcn_wmma_f32_16x16x32_f16(false, a2, false, b, (short)0, acc2, false, false);
    acc3 = __builtin_amdgcn_wmma_f32_16x16x32_f16(false, a3, false, b, (short)0, acc3, false, false);
  }

  const float bv = bias[n];
  auto store_tile = [&](const v8f& acc, int mT) {
#pragma unroll
    for (int r = 0; r < 8; ++r) {
      int m = mT + r + 8 * hi;            // C/D layout: VGPR r -> row r (+8 for hi lanes)
      float v = acc[r] + bv;
      v = v > 0.0f ? v : 0.0f;
      if (Yh) Yh[(size_t)m * Cout + n] = (_Float16)v;
      else    Yf[(size_t)m * Cout + n] = v;
    }
  };
  store_tile(acc0, mBase +  0);
  store_tile(acc1, mBase + 16);
  store_tile(acc2, mBase + 32);
  store_tile(acc3, mBase + 48);
}

// ------------------------------------------------------------------
__global__ void maxpool_kernel(const _Float16* __restrict__ Y, int nsample, int C,
                               int total_bs, _Float16* __restrict__ out) {
  int tid = blockIdx.x * blockDim.x + threadIdx.x;
  if (tid >= total_bs * C) return;
  int c  = tid % C;
  int bs = tid / C;
  float m = -3.0e38f;
  const _Float16* base = Y + ((size_t)bs * nsample) * C + c;
  for (int k = 0; k < nsample; ++k) {
    float v = (float)base[(size_t)k * C];
    m = v > m ? v : m;
  }
  out[(size_t)bs * C + c] = (_Float16)m;
}

// ------------------------------------------------------------------
// Feature propagation: 3-NN inverse (squared-)distance interpolation,
// then X row = concat(feats1, interp). One thread per unknown point.
// ------------------------------------------------------------------
__global__ void fp_build_kernel(const float* __restrict__ unknown, const float* __restrict__ known,
                                const _Float16* __restrict__ feats1, int C1,
                                const _Float16* __restrict__ feats2, int C2,
                                int n_unk, int n_kn, _Float16* __restrict__ X) {
  int tid = blockIdx.x * blockDim.x + threadIdx.x;
  if (tid >= BATCH * n_unk) return;
  int b = tid / n_unk;
  const float* u  = unknown + (size_t)tid * 3;
  const float* kp = known + (size_t)b * n_kn * 3;
  float d0 = 1e30f, d1 = 1e30f, d2 = 1e30f;
  int   i0 = 0,     i1 = 0,     i2 = 0;
  for (int j = 0; j < n_kn; ++j) {
    float dx = kp[j*3+0]-u[0], dy = kp[j*3+1]-u[1], dz = kp[j*3+2]-u[2];
    float d = dx*dx + dy*dy + dz*dz;
    if (d < d0)      { d2 = d1; i2 = i1; d1 = d0; i1 = i0; d0 = d; i0 = j; }
    else if (d < d1) { d2 = d1; i2 = i1; d1 = d; i1 = j; }
    else if (d < d2) { d2 = d; i2 = j; }
  }
  float w0 = 1.0f / (fmaxf(d0, 0.0f) + 1e-8f);
  float w1 = 1.0f / (fmaxf(d1, 0.0f) + 1e-8f);
  float w2 = 1.0f / (fmaxf(d2, 0.0f) + 1e-8f);
  float wsum = w0 + w1 + w2;
  w0 /= wsum; w1 /= wsum; w2 /= wsum;
  _Float16* xr = X + (size_t)tid * (C1 + C2);
  const _Float16* f1r = feats1 + (size_t)tid * C1;
  for (int c = 0; c < C1; ++c) xr[c] = f1r[c];
  const _Float16* f2b = feats2 + (size_t)b * n_kn * C2;
  for (int c = 0; c < C2; ++c) {
    float v = w0 * (float)f2b[(size_t)i0 * C2 + c] +
              w1 * (float)f2b[(size_t)i1 * C2 + c] +
              w2 * (float)f2b[(size_t)i2 * C2 + c];
    xr[C1 + c] = (_Float16)v;
  }
}

// ------------------------------------------------------------------
__global__ void copy_f32_kernel(const float* __restrict__ src, float* __restrict__ dst, int n) {
  int tid = blockIdx.x * blockDim.x + threadIdx.x;
  if (tid < n) dst[tid] = src[tid];
}

// ------------------------------------------------------------------
extern "C" void kernel_launch(void* const* d_in, const int* in_sizes, int n_in,
                              void* d_out, int out_size, void* d_ws, size_t ws_size,
                              hipStream_t stream) {
  (void)in_sizes; (void)n_in; (void)out_size; (void)ws_size;
  const float* pc     = (const float*)d_in[0];
  const float* sa1_w0 = (const float*)d_in[1];  const float* sa1_b0 = (const float*)d_in[2];
  const float* sa1_w1 = (const float*)d_in[3];  const float* sa1_b1 = (const float*)d_in[4];
  const float* sa1_w2 = (const float*)d_in[5];  const float* sa1_b2 = (const float*)d_in[6];
  const float* sa2_w0 = (const float*)d_in[7];  const float* sa2_b0 = (const float*)d_in[8];
  const float* sa2_w1 = (const float*)d_in[9];  const float* sa2_b1 = (const float*)d_in[10];
  const float* sa3_w0 = (const float*)d_in[11]; const float* sa3_b0 = (const float*)d_in[12];
  const float* sa3_w1 = (const float*)d_in[13]; const float* sa3_b1 = (const float*)d_in[14];
  const float* sa4_w0 = (const float*)d_in[15]; const float* sa4_b0 = (const float*)d_in[16];
  const float* sa4_w1 = (const float*)d_in[17]; const float* sa4_b1 = (const float*)d_in[18];
  const float* fp1_w0 = (const float*)d_in[19]; const float* fp1_b0 = (const float*)d_in[20];
  const float* fp1_w1 = (const float*)d_in[21]; const float* fp1_b1 = (const float*)d_in[22];
  const float* fp2_w0 = (const float*)d_in[23]; const float* fp2_b0 = (const float*)d_in[24];
  const float* fp2_w1 = (const float*)d_in[25]; const float* fp2_b1 = (const float*)d_in[26];

  const int N0 = 32768;
  char* ws = (char*)d_ws;
  size_t off = 0;
  auto alloc = [&](size_t bytes) -> void* {
    void* p = ws + off;
    off = (off + bytes + 255) & ~(size_t)255;
    return p;
  };
  float*     mind   = (float*)alloc((size_t)BATCH * N0 * 4);
  int*       inds1  = (int*)alloc((size_t)BATCH * 512 * 4);
  int*       inds2  = (int*)alloc((size_t)BATCH * 256 * 4);
  int*       inds3  = (int*)alloc((size_t)BATCH * 64 * 4);
  int*       inds4  = (int*)alloc((size_t)BATCH * 16 * 4);
  float*     xyz1   = (float*)alloc((size_t)BATCH * 512 * 3 * 4);
  float*     xyz2   = (float*)alloc((size_t)BATCH * 256 * 3 * 4);
  float*     xyz3   = (float*)alloc((size_t)BATCH * 64 * 3 * 4);
  float*     xyz4   = (float*)alloc((size_t)BATCH * 16 * 3 * 4);
  int*       idx1   = (int*)alloc((size_t)BATCH * 512 * 64 * 4);
  int*       idx2   = (int*)alloc((size_t)BATCH * 256 * 32 * 4);
  int*       idx3   = (int*)alloc((size_t)BATCH * 64 * 16 * 4);
  int*       idx4   = (int*)alloc((size_t)BATCH * 16 * 16 * 4);
  _Float16*  f1     = (_Float16*)alloc((size_t)BATCH * 512 * 64 * 2);
  _Float16*  f2     = (_Float16*)alloc((size_t)BATCH * 256 * 128 * 2);
  _Float16*  f3     = (_Float16*)alloc((size_t)BATCH * 64 * 256 * 2);
  _Float16*  f4     = (_Float16*)alloc((size_t)BATCH * 16 * 256 * 2);
  _Float16*  ffp1   = (_Float16*)alloc((size_t)BATCH * 64 * 128 * 2);
  _Float16*  wpack  = (_Float16*)alloc((size_t)512 * 1024);          // 512 KB weight pack
  _Float16*  bufA   = (_Float16*)alloc((size_t)16 << 20);
  _Float16*  bufB   = (_Float16*)alloc((size_t)16 << 20);

  auto cdiv = [](int a, int b) { return (a + b - 1) / b; };
  auto gemm = [&](const _Float16* Xb, int M, int Kpad, int Cin, const float* W, const float* bias,
                  int Cout, _Float16* Yh, float* Yf) {
    pack_weights_kernel<<<cdiv(Kpad * Cout, 256), 256, 0, stream>>>(W, Cin, Cout, Kpad, wpack);
    dim3 grid(cdiv(M, 16 * GEMM_MB * 4), Cout / 16);   // 4 waves/block
    gemm_bias_relu_wmma<<<grid, 128, 0, stream>>>(Xb, M, Kpad, wpack, bias, Cout, Yh, Yf);
  };

  // ---------------- SA1: 32768 -> 512, r=0.1, ns=64, MLP 3->32->32->64 ----
  fps_kernel<<<BATCH, FPS_THREADS, 0, stream>>>(pc, N0, 512, inds1, mind);
  gather_xyz_kernel<<<cdiv(BATCH*512, 256), 256, 0, stream>>>(pc, inds1, xyz1, 512, N0);
  ball_query_kernel<<<cdiv(BATCH*512, 128), 128, 0, stream>>>(pc, xyz1, N0, 512, 0.01f, 64, idx1);
  group_kernel<<<cdiv(BATCH*512*64, 256), 256, 0, stream>>>(pc, xyz1, nullptr, 0, N0,
                                                            idx1, 512, 64, 10.0f, 32, bufA);
  gemm(bufA, BATCH*512*64, 32, 3,  sa1_w0, sa1_b0, 32, bufB, nullptr);
  gemm(bufB, BATCH*512*64, 32, 32, sa1_w1, sa1_b1, 32, bufA, nullptr);
  gemm(bufA, BATCH*512*64, 32, 32, sa1_w2, sa1_b2, 64, bufB, nullptr);
  maxpool_kernel<<<cdiv(BATCH*512*64, 256), 256, 0, stream>>>(bufB, 64, 64, BATCH*512, f1);

  // ---------------- SA2: 512 -> 256, r=0.2, ns=32, MLP 67->64->128 --------
  fps_kernel<<<BATCH, FPS_THREADS, 0, stream>>>(xyz1, 512, 256, inds2, mind);
  gather_xyz_kernel<<<cdiv(BATCH*256, 256), 256, 0, stream>>>(xyz1, inds2, xyz2, 256, 512);
  ball_query_kernel<<<cdiv(BATCH*256, 128), 128, 0, stream>>>(xyz1, xyz2, 512, 256, 0.04f, 32, idx2);
  group_kernel<<<cdiv(BATCH*256*32, 256), 256, 0, stream>>>(xyz1, xyz2, f1, 64, 512,
                                                            idx2, 256, 32, 5.0f, 96, bufA);
  gemm(bufA, BATCH*256*32, 96, 67, sa2_w0, sa2_b0, 64,  bufB, nullptr);
  gemm(bufB, BATCH*256*32, 64, 64, sa2_w1, sa2_b1, 128, bufA, nullptr);
  maxpool_kernel<<<cdiv(BATCH*256*128, 256), 256, 0, stream>>>(bufA, 32, 128, BATCH*256, f2);

  // ---------------- SA3: 256 -> 64, r=0.4, ns=16, MLP 131->128->256 -------
  fps_kernel<<<BATCH, FPS_THREADS, 0, stream>>>(xyz2, 256, 64, inds3, mind);
  gather_xyz_kernel<<<cdiv(BATCH*64, 256), 256, 0, stream>>>(xyz2, inds3, xyz3, 64, 256);
  ball_query_kernel<<<cdiv(BATCH*64, 128), 128, 0, stream>>>(xyz2, xyz3, 256, 64, 0.16f, 16, idx3);
  group_kernel<<<cdiv(BATCH*64*16, 256), 256, 0, stream>>>(xyz2, xyz3, f2, 128, 256,
                                                           idx3, 64, 16, 2.5f, 160, bufA);
  gemm(bufA, BATCH*64*16, 160, 131, sa3_w0, sa3_b0, 128, bufB, nullptr);
  gemm(bufB, BATCH*64*16, 128, 128, sa3_w1, sa3_b1, 256, bufA, nullptr);
  maxpool_kernel<<<cdiv(BATCH*64*256, 256), 256, 0, stream>>>(bufA, 16, 256, BATCH*64, f3);

  // ---------------- SA4: 64 -> 16, r=0.8, ns=16, MLP 259->256->256 --------
  fps_kernel<<<BATCH, FPS_THREADS, 0, stream>>>(xyz3, 64, 16, inds4, mind);
  gather_xyz_kernel<<<cdiv(BATCH*16, 256), 256, 0, stream>>>(xyz3, inds4, xyz4, 16, 64);
  ball_query_kernel<<<cdiv(BATCH*16, 128), 128, 0, stream>>>(xyz3, xyz4, 64, 16, 0.64f, 16, idx4);
  group_kernel<<<cdiv(BATCH*16*16, 256), 256, 0, stream>>>(xyz3, xyz4, f3, 256, 64,
                                                           idx4, 16, 16, 1.25f, 288, bufA);
  gemm(bufA, BATCH*16*16, 288, 259, sa4_w0, sa4_b0, 256, bufB, nullptr);
  gemm(bufB, BATCH*16*16, 256, 256, sa4_w1, sa4_b1, 256, bufA, nullptr);
  maxpool_kernel<<<cdiv(BATCH*16*256, 256), 256, 0, stream>>>(bufA, 16, 256, BATCH*16, f4);

  // ---------------- FP1: xyz3(64) <- xyz4(16), MLP 512->256->128 ----------
  fp_build_kernel<<<cdiv(BATCH*64, 64), 64, 0, stream>>>(xyz3, xyz4, f3, 256, f4, 256,
                                                         64, 16, bufA);
  gemm(bufA, BATCH*64, 512, 512, fp1_w0, fp1_b0, 256, bufB, nullptr);
  gemm(bufB, BATCH*64, 256, 256, fp1_w1, fp1_b1, 128, ffp1, nullptr);

  // ---------------- FP2: xyz2(256) <- xyz3(64), MLP 256->128->64 ----------
  fp_build_kernel<<<cdiv(BATCH*256, 64), 64, 0, stream>>>(xyz2, xyz3, f2, 128, ffp1, 128,
                                                          256, 64, bufA);
  gemm(bufA, BATCH*256, 256, 256, fp2_w0, fp2_b0, 128, bufB, nullptr);
  gemm(bufB, BATCH*256, 128, 128, fp2_w1, fp2_b1, 64, nullptr, (float*)d_out);

  // ---------------- output tail: xyz2 (4,256,3) ---------------------------
  copy_f32_kernel<<<cdiv(BATCH*256*3, 256), 256, 0, stream>>>(xyz2,
      (float*)d_out + (size_t)BATCH*256*64, BATCH*256*3);
}
